// Recce_76261439307880
// MI455X (gfx1250) — matrix operand order repaired and checked by reading, not compiled, not run on hardware
//
#include <hip/hip_runtime.h>

// ---------------------------------------------------------------------------
// MI455X (gfx1250, wave32) implementation of the graph-reasoning block.
// Large GEMMs: v_wmma_f32_16x16x32_f16 fed by double-buffered LDS tiles
// filled with GLOBAL_LOAD_ASYNC_TO_LDS_B128 (ASYNCcnt-tracked DMA).
// ---------------------------------------------------------------------------

typedef _Float16 h16;
typedef __attribute__((ext_vector_type(16))) _Float16 v16h;
typedef __attribute__((ext_vector_type(8)))  _Float16 v8h;
typedef __attribute__((ext_vector_type(8)))  float    v8f;
typedef __attribute__((ext_vector_type(4)))  int      v4i;

#define LDS_STRIDE 48   // halves per LDS row: 32 data + 16 pad (96B stride, 32B-aligned frags)

#if defined(__has_builtin)
#if __has_builtin(__builtin_amdgcn_global_load_async_to_lds_b128)
#define HAVE_ASYNC_BUILTIN 1
#endif
#if __has_builtin(__builtin_amdgcn_s_wait_asynccnt)
#define HAVE_WAIT_ASYNC_BUILTIN 1
#endif
#endif

__device__ __forceinline__ void wait_async0() {
#if defined(HAVE_WAIT_ASYNC_BUILTIN)
  __builtin_amdgcn_s_wait_asynccnt(0);
#else
  asm volatile("s_wait_asynccnt 0x0" ::: "memory");
#endif
}

// Copy 32 bytes (16 halves) per lane: global -> LDS, asynchronously.
// INST_OFFSET is applied to both the global and the LDS address (ISA 10.x),
// so two b128 ops at offset 0 and 16 move one contiguous 32B chunk.
// Builtin signature (from hipcc diagnostics): (int4 AS1*, int4 AS3*, Imm, Imm).
__device__ __forceinline__ void async_cp32(const h16* g, h16* l) {
#if defined(HAVE_ASYNC_BUILTIN)
  __builtin_amdgcn_global_load_async_to_lds_b128(
      (__attribute__((address_space(1))) v4i*)g,
      (__attribute__((address_space(3))) v4i*)l, 0, 0);
  __builtin_amdgcn_global_load_async_to_lds_b128(
      (__attribute__((address_space(1))) v4i*)g,
      (__attribute__((address_space(3))) v4i*)l, 16, 0);
#else
  unsigned laddr =
      (unsigned)(unsigned long long)(__attribute__((address_space(3))) h16*)l;
  unsigned long long gaddr = (unsigned long long)g;
  asm volatile("global_load_async_to_lds_b128 %0, %1, off\n\t"
               "global_load_async_to_lds_b128 %0, %1, off offset:16"
               :: "v"(laddr), "v"(gaddr) : "memory");
#endif
}

__device__ __forceinline__ float sigmoidf_(float x) {
  return 1.0f / (1.0f + __expf(-x));
}

// ---------------------------------------------------------------------------
// Generic batched WMMA GEMM:
//   D[m,n] = epilogue( sum_k A[m,k] * Bt[n,k] )     (per batch z)
//   A : f16 [Mpad, Kp], weights, shared across batch (Mpad = 128*gridDim.y)
//   Bt: f16 [N, Kp]    per batch (N = 128*gridDim.x), "activations transposed"
// Output:
//   OUTF32==0 : f16 transposed  Ct[n*ldc + m]       (ready as B of next GEMM)
//   OUTF32==1 : f32 row-major   C[m*ldc + n], only rows m < m_valid
// EPI: 0=none 1=relu 3=1-sigmoid
// Block: 256 threads = 8 waves; block tile 128x128, BK=32; wave tile 32x64.
// Double-buffered LDS, async DMA fill overlapped with WMMA.
// ---------------------------------------------------------------------------
template<int EPI, int OUTF32>
__global__ __launch_bounds__(256)
void gemm_wmma_k(const h16* __restrict__ A,
                 const h16* __restrict__ Bt,
                 void* __restrict__ Cout,
                 int Kp, size_t strideB, size_t strideC,
                 int ldc, int m_valid)
{
  __shared__ h16 As[2][128 * LDS_STRIDE];
  __shared__ h16 Bs[2][128 * LDS_STRIDE];

  const int tid   = threadIdx.x;
  const int lane  = tid & 31;
  const int wave  = tid >> 5;
  const int wm0   = (wave >> 1) * 32;   // wave row offset   (4 wave-rows)
  const int wn0   = (wave & 1) * 64;    // wave col offset   (2 wave-cols)
  const int lmod  = lane & 15;
  const int hiH   = (lane >> 4);        // 0 for lanes 0-15, 1 for 16-31

  const int mBlk = blockIdx.y * 128;
  const int nBlk = blockIdx.x * 128;
  const h16* Bb  = Bt + (size_t)blockIdx.z * strideB;

  // staging: 256 threads move 128 rows x 32 halves (32B per thread per tile)
  const int srow = tid >> 1;
  const int scol = (tid & 1) * 16;
  const int soff = srow * LDS_STRIDE + scol;
  const h16* gA = A  + (size_t)(mBlk + srow) * Kp + scol;
  const h16* gB = Bb + (size_t)(nBlk + srow) * Kp + scol;

  // A-fragment per-lane K chunks (matching B's contiguous order):
  //   lanes 0-15 : K = 0..15;  lanes 16-31 : K = 16..31
  const int aC0 = hiH * 16;
  // B-fragment per-lane K chunk (dense 32x16 B): lanes 0-15 K=0..15, 16-31 K=16..31
  const int bC0 = hiH * 16;

  v8f acc[2][4] = {};

  const int nTiles = Kp >> 5;

  // prologue: DMA tile 0 into buffer 0
  async_cp32(gA, &As[0][soff]);
  async_cp32(gB, &Bs[0][soff]);
  wait_async0();
  __syncthreads();

  for (int kt = 0; kt < nTiles; ++kt) {
    const int cur = kt & 1;

    // overlap: DMA next tile into the other buffer while computing this one
    if (kt + 1 < nTiles) {
      async_cp32(gA + (size_t)(kt + 1) * 32, &As[cur ^ 1][soff]);
      async_cp32(gB + (size_t)(kt + 1) * 32, &Bs[cur ^ 1][soff]);
    }

    v16h a[2];
#pragma unroll
    for (int wm = 0; wm < 2; ++wm) {
      const h16* r = &As[cur][(wm0 + wm * 16 + lmod) * LDS_STRIDE];
      v8h lo = *(const v8h*)&r[aC0];
      v8h hi = *(const v8h*)&r[aC0 + 8];
      a[wm] = __builtin_shufflevector(lo, hi, 0,1,2,3,4,5,6,7,8,9,10,11,12,13,14,15);
    }
    v16h b[4];
#pragma unroll
    for (int wn = 0; wn < 4; ++wn)
      b[wn] = *(const v16h*)&Bs[cur][(wn0 + wn * 16 + lmod) * LDS_STRIDE + bC0];

#pragma unroll
    for (int wm = 0; wm < 2; ++wm)
#pragma unroll
      for (int wn = 0; wn < 4; ++wn)
        acc[wm][wn] = __builtin_amdgcn_wmma_f32_16x16x32_f16(
            false, a[wm], false, b[wn], (short)0, acc[wm][wn], false, false);

    wait_async0();      // our next-tile DMA has landed in LDS
    __syncthreads();    // everyone's landed; everyone done reading cur
  }

  // D layout: VGPR r -> M = r (+8 for lanes 16-31); lane&15 -> N
  const int mW = mBlk + wm0 + hiH * 8;
  const int nW = nBlk + wn0 + lmod;
#pragma unroll
  for (int wm = 0; wm < 2; ++wm) {
#pragma unroll
    for (int wn = 0; wn < 4; ++wn) {
#pragma unroll
      for (int r = 0; r < 8; ++r) {
        float v = acc[wm][wn][r];
        if (EPI == 1) v = fmaxf(v, 0.0f);
        else if (EPI == 3) v = 1.0f - sigmoidf_(v);
        const int m = mW + wm * 16 + r;
        const int n = nW + wn * 16;
        if (OUTF32) {
          if (m < m_valid) {
            float* C = (float*)Cout + (size_t)blockIdx.z * strideC;
            C[(size_t)m * ldc + n] = v;
          }
        } else {
          h16* C = (h16*)Cout + (size_t)blockIdx.z * strideC;
          C[(size_t)n * ldc + m] = (h16)v;
        }
      }
    }
  }
}

// ---------------------------------------------------------------------------
// Packing kernels
// ---------------------------------------------------------------------------
__global__ void pack_w_k(const float* __restrict__ W, h16* __restrict__ Wp,
                         int O, int I, int Op, int Ip)
{
  int idx = blockIdx.x * blockDim.x + threadIdx.x;
  if (idx >= Op * Ip) return;
  int o = idx / Ip, i = idx - o * Ip;
  Wp[idx] = (o < O && i < I) ? (h16)W[o * I + i] : (h16)0.0f;
}

// vert_a [64][728][256] -> XaT f16 [64][256 n][768] (K padded with zeros)
__global__ void pack_xa_k(const float* __restrict__ va, h16* __restrict__ XaT)
{
  size_t idx = (size_t)blockIdx.x * blockDim.x + threadIdx.x;
  int i = (int)(idx % 768);
  size_t r = idx / 768;
  int n = (int)(r % 256);
  int b = (int)(r / 256);
  XaT[idx] = (i < 728) ? (h16)va[((size_t)b * 728 + i) * 256 + n] : (h16)0.0f;
}

// vert_b [64][256][32][32] -> pbT f16 [64][k*256+n][256], k=dr*2+dc, n=h*16+w
__global__ void pack_pb_k(const float* __restrict__ vb, h16* __restrict__ pbT)
{
  size_t idx = (size_t)blockIdx.x * blockDim.x + threadIdx.x;
  int c = (int)(idx % 256);
  size_t r = idx / 256;
  int row = (int)(r % 1024);
  int b = (int)(r / 1024);
  int k = row >> 8, n = row & 255;
  int dr = k >> 1, dc = k & 1, hh = n >> 4, ww = n & 15;
  pbT[idx] = (h16)vb[(((size_t)b * 256 + c) * 32 + (hh * 2 + dr)) * 32 + (ww * 2 + dc)];
}

// vert_c [64][128][64][64] -> pcT f16 [64][k*256+n][128], k=dr*4+dc
__global__ void pack_pc_k(const float* __restrict__ vc, h16* __restrict__ pcT)
{
  size_t idx = (size_t)blockIdx.x * blockDim.x + threadIdx.x;
  int c = (int)(idx % 128);
  size_t r = idx / 128;
  int row = (int)(r % 4096);
  int b = (int)(r / 4096);
  int k = row >> 8, n = row & 255;
  int dr = k >> 2, dc = k & 3, hh = n >> 4, ww = n & 15;
  pcT[idx] = (h16)vc[(((size_t)b * 128 + c) * 64 + (hh * 4 + dr)) * 64 + (ww * 4 + dc)];
}

// ---------------------------------------------------------------------------
// Aggregation: logits -> relu -> softmax over k -> weighted sum -> gate ->
// sigmoid, concatenated with sigmoid(emb_a) into aggT [64][256 n][768].
// One block per (b, n); 256 threads = one per channel d.
// ---------------------------------------------------------------------------
__global__ __launch_bounds__(256)
void agg_k(const h16* __restrict__ embaT, const h16* __restrict__ embbT,
           const h16* __restrict__ embcT, const h16* __restrict__ gatebT,
           const h16* __restrict__ gatecT, const float* __restrict__ rab,
           const float* __restrict__ rac, h16* __restrict__ aggT)
{
  const int n = blockIdx.x;
  const int b = blockIdx.y;
  const int t = threadIdx.x;

  __shared__ float red[256];
  __shared__ float logb[4];
  __shared__ float logc[16];

  const size_t baseA = ((size_t)b * 256 + n) * 256;
  const float ea = (float)embaT[baseA + t];

  auto reduceAll = [&](float v) -> float {
    red[t] = v;
    __syncthreads();
    for (int s = 128; s > 0; s >>= 1) {
      if (t < s) red[t] += red[t + s];
      __syncthreads();
    }
    float r = red[0];
    __syncthreads();
    return r;
  };

  const float la_b = reduceAll(ea * rab[t]);
  const float la_c = reduceAll(ea * rac[t]);

  float vb[4];
#pragma unroll
  for (int k = 0; k < 4; ++k) {
    vb[k] = (float)embbT[((size_t)b * 1024 + (size_t)k * 256 + n) * 256 + t];
    float s = reduceAll(vb[k] * rab[256 + t]);
    if (t == 0) logb[k] = fmaxf(la_b + s, 0.0f);
  }
  float vc[16];
#pragma unroll
  for (int k = 0; k < 16; ++k) {
    vc[k] = (float)embcT[((size_t)b * 4096 + (size_t)k * 256 + n) * 256 + t];
    float s = reduceAll(vc[k] * rac[256 + t]);
    if (t == 0) logc[k] = fmaxf(la_c + s, 0.0f);
  }
  __syncthreads();

  // tiny softmaxes computed redundantly per thread
  float mb = logb[0];
#pragma unroll
  for (int k = 1; k < 4; ++k) mb = fmaxf(mb, logb[k]);
  float sb = 0.0f, eb[4];
#pragma unroll
  for (int k = 0; k < 4; ++k) { eb[k] = __expf(logb[k] - mb); sb += eb[k]; }
  float aggb = 0.0f;
#pragma unroll
  for (int k = 0; k < 4; ++k) aggb += (eb[k] / sb) * vb[k];

  float mc = logc[0];
#pragma unroll
  for (int k = 1; k < 16; ++k) mc = fmaxf(mc, logc[k]);
  float sc = 0.0f, ec[16];
#pragma unroll
  for (int k = 0; k < 16; ++k) { ec[k] = __expf(logc[k] - mc); sc += ec[k]; }
  float aggc = 0.0f;
#pragma unroll
  for (int k = 0; k < 16; ++k) aggc += (ec[k] / sc) * vc[k];

  const float gb = (float)gatebT[baseA + t];
  const float gc = (float)gatecT[baseA + t];

  h16* outp = aggT + ((size_t)b * 256 + n) * 768;
  outp[t]       = (h16)sigmoidf_(aggb * gb);   // channels   0..255 : agg_b
  outp[256 + t] = (h16)sigmoidf_(aggc * gc);   // channels 256..511 : agg_c
  outp[512 + t] = (h16)sigmoidf_(ea);          // channels 512..767 : emb_a
}

// ---------------------------------------------------------------------------
// Launch
// ---------------------------------------------------------------------------
extern "C" void kernel_launch(void* const* d_in, const int* in_sizes, int n_in,
                              void* d_out, int out_size, void* d_ws, size_t ws_size,
                              hipStream_t stream)
{
  (void)in_sizes; (void)n_in; (void)out_size; (void)ws_size;

  const float* vert_a = (const float*)d_in[0];
  const float* vert_b = (const float*)d_in[1];
  const float* vert_c = (const float*)d_in[2];
  const float* va_w1  = (const float*)d_in[3];
  const float* va_w2  = (const float*)d_in[4];
  const float* gb_w   = (const float*)d_in[5];
  const float* gc_w   = (const float*)d_in[6];
  const float* vb_w1  = (const float*)d_in[7];
  const float* vb_w2  = (const float*)d_in[8];
  const float* vc_w1  = (const float*)d_in[9];
  const float* vc_w2  = (const float*)d_in[10];
  const float* rab_w  = (const float*)d_in[11];
  const float* rac_w  = (const float*)d_in[12];
  const float* rp_w1  = (const float*)d_in[13];
  const float* rp_w2  = (const float*)d_in[14];
  float* out = (float*)d_out;

  h16* base = (h16*)d_ws;
  size_t off = 0;
  auto alloc = [&](size_t nelem) {
    h16* p = base + off;
    off += (nelem + 127) & ~(size_t)127;   // 256B-aligned chunks
    return p;
  };

  h16* va_w1p = alloc((size_t)256 * 768);
  h16* va_w2p = alloc((size_t)256 * 256);
  h16* gb_wp  = alloc((size_t)256 * 768);
  h16* gc_wp  = alloc((size_t)256 * 768);
  h16* vb_w1p = alloc((size_t)256 * 256);
  h16* vb_w2p = alloc((size_t)256 * 256);
  h16* vc_w1p = alloc((size_t)256 * 128);
  h16* vc_w2p = alloc((size_t)256 * 256);
  h16* rp_w1p = alloc((size_t)768 * 768);
  h16* rp_w2p = alloc((size_t)768 * 768);

  h16* XaT    = alloc((size_t)64 * 256 * 768);
  h16* t1T    = alloc((size_t)64 * 256 * 256);
  h16* embaT  = alloc((size_t)64 * 256 * 256);
  h16* gatebT = alloc((size_t)64 * 256 * 256);
  h16* gatecT = alloc((size_t)64 * 256 * 256);
  h16* pbT    = alloc((size_t)64 * 1024 * 256);
  h16* h1bT   = alloc((size_t)64 * 1024 * 256);
  h16* embbT  = alloc((size_t)64 * 1024 * 256);
  h16* pcT    = alloc((size_t)64 * 4096 * 128);
  h16* h1cT   = alloc((size_t)64 * 4096 * 256);
  h16* embcT  = alloc((size_t)64 * 4096 * 256);
  h16* aggT   = alloc((size_t)64 * 256 * 768);
  h16* zT     = alloc((size_t)64 * 256 * 768);

  // ---- weight packing (f32 -> f16, zero-padded) ----
  auto PW = [&](const float* src, h16* dst, int O, int I, int Op, int Ip) {
    int tot = Op * Ip;
    pack_w_k<<<(tot + 255) / 256, 256, 0, stream>>>(src, dst, O, I, Op, Ip);
  };
  PW(va_w1, va_w1p, 256, 728, 256, 768);
  PW(va_w2, va_w2p, 256, 256, 256, 256);
  PW(gb_w,  gb_wp,  256, 728, 256, 768);
  PW(gc_w,  gc_wp,  256, 728, 256, 768);
  PW(vb_w1, vb_w1p, 256, 256, 256, 256);
  PW(vb_w2, vb_w2p, 256, 256, 256, 256);
  PW(vc_w1, vc_w1p, 256, 128, 256, 128);
  PW(vc_w2, vc_w2p, 256, 256, 256, 256);
  PW(rp_w1, rp_w1p, 728, 768, 768, 768);
  PW(rp_w2, rp_w2p, 728, 728, 768, 768);

  // ---- activation packing (f32 -> f16 transposed / unfolded) ----
  {
    size_t tot = (size_t)64 * 256 * 768;
    pack_xa_k<<<(unsigned)(tot / 256), 256, 0, stream>>>(vert_a, XaT);
  }
  {
    size_t tot = (size_t)64 * 1024 * 256;
    pack_pb_k<<<(unsigned)(tot / 256), 256, 0, stream>>>(vert_b, pbT);
  }
  {
    size_t tot = (size_t)64 * 4096 * 128;
    pack_pc_k<<<(unsigned)(tot / 256), 256, 0, stream>>>(vert_c, pcT);
  }

  const dim3 blk(256);

  // ---- emb_a path: t1 = relu(va_w1 @ Xa); emb_a = va_w2 @ t1 ----
  gemm_wmma_k<1, 0><<<dim3(2, 2, 64), blk, 0, stream>>>(
      va_w1p, XaT, t1T, 768, (size_t)256 * 768, (size_t)256 * 256, 256, 256);
  gemm_wmma_k<0, 0><<<dim3(2, 2, 64), blk, 0, stream>>>(
      va_w2p, t1T, embaT, 256, (size_t)256 * 256, (size_t)256 * 256, 256, 256);

  // ---- gates: 1 - sigmoid(g* @ Xa) ----
  gemm_wmma_k<3, 0><<<dim3(2, 2, 64), blk, 0, stream>>>(
      gb_wp, XaT, gatebT, 768, (size_t)256 * 768, (size_t)256 * 256, 256, 256);
  gemm_wmma_k<3, 0><<<dim3(2, 2, 64), blk, 0, stream>>>(
      gc_wp, XaT, gatecT, 768, (size_t)256 * 768, (size_t)256 * 256, 256, 256);

  // ---- emb_b: relu(pb @ vb_w1.T) @ vb_w2.T   (N = 4*256 = 1024 rows/batch) ----
  gemm_wmma_k<1, 0><<<dim3(8, 2, 64), blk, 0, stream>>>(
      vb_w1p, pbT, h1bT, 256, (size_t)1024 * 256, (size_t)1024 * 256, 256, 256);
  gemm_wmma_k<0, 0><<<dim3(8, 2, 64), blk, 0, stream>>>(
      vb_w2p, h1bT, embbT, 256, (size_t)1024 * 256, (size_t)1024 * 256, 256, 256);

  // ---- emb_c: relu(pc @ vc_w1.T) @ vc_w2.T   (N = 16*256 = 4096 rows/batch) ----
  gemm_wmma_k<1, 0><<<dim3(32, 2, 64), blk, 0, stream>>>(
      vc_w1p, pcT, h1cT, 128, (size_t)4096 * 128, (size_t)4096 * 256, 256, 256);
  gemm_wmma_k<0, 0><<<dim3(32, 2, 64), blk, 0, stream>>>(
      vc_w2p, h1cT, embcT, 256, (size_t)4096 * 256, (size_t)4096 * 256, 256, 256);

  // ---- aggregation + concat + sigmoid -> aggT [64][256][768] ----
  agg_k<<<dim3(256, 64), blk, 0, stream>>>(
      embaT, embbT, embcT, gatebT, gatecT, rab_w, rac_w, aggT);

  // ---- final: out = rp_w2 @ relu(rp_w1 @ agg_abc) ----
  gemm_wmma_k<1, 0><<<dim3(2, 6, 64), blk, 0, stream>>>(
      rp_w1p, aggT, zT, 768, (size_t)256 * 768, (size_t)256 * 768, 768, 768);
  gemm_wmma_k<0, 1><<<dim3(2, 6, 64), blk, 0, stream>>>(
      rp_w2p, zT, out, 768, (size_t)256 * 768, (size_t)728 * 256, 256, 728);
}
